// TinyGCN_19327352832217
// MI455X (gfx1250) — matrix operand order repaired and checked by reading, not compiled, not run on hardware
//
#include <hip/hip_runtime.h>

// ---------------------------------------------------------------------------
// TinyGCN forward for MI455X (gfx1250, wave32, WMMA)
//   h      = X @ W_gcn.T                      (bf16 WMMA, f32 accum, 2x4 blocked)
//   agg    = scatter-add over edges of h[src]*dinv[src]*dinv[dst]  (L2 atomics)
//   hidden = relu(agg + h*dinv^2 + b_gcn)     (self-loop fused here)
//   out    = hidden @ W_cls.T + b_cls         (bf16 WMMA, f32 accum, 2x4 blocked)
// ---------------------------------------------------------------------------

#define N_NODES   50000
#define N_EDGES   800000
#define IN_DIM    512
#define HID_DIM   512
#define N_CLASSES 64

typedef __bf16        v16bf __attribute__((ext_vector_type(16)));
typedef float         v8f   __attribute__((ext_vector_type(8)));
typedef unsigned int  v4u   __attribute__((ext_vector_type(4)));
typedef unsigned int  v8u   __attribute__((ext_vector_type(8)));

// float -> bf16 bits, round-to-nearest-even
__device__ __forceinline__ unsigned short f2bf(float f) {
    unsigned int u = __float_as_uint(f);
    u += 0x7FFFu + ((u >> 16) & 1u);
    return (unsigned short)(u >> 16);
}

// ---- WMMA fragment loaders (A and B both use contiguous k order k = 16h + s) ----
// A 16x32 bf16 tile, row-major source: lane half 0 -> K 0..15,
// half 1 -> K 16..31; two contiguous 16B chunks per lane.
__device__ __forceinline__ v16bf load_afrag(const unsigned short* __restrict__ p,
                                            int k0, int half) {
    v4u a0 = *(const v4u*)(p + k0 +      half * 16);
    v4u a1 = *(const v4u*)(p + k0 + 8 + half * 16);
    return __builtin_bit_cast(
        v16bf, __builtin_shufflevector(a0, a1, 0, 1, 2, 3, 4, 5, 6, 7));
}
// B 32x16 bf16 tile with B[k][j] = W[j][k]: lane half 0 -> K 0..15,
// half 1 -> K 16..31; one contiguous 32B chunk of a W row per lane.
__device__ __forceinline__ v16bf load_bfrag(const unsigned short* __restrict__ p,
                                            int k0, int half) {
    return __builtin_bit_cast(v16bf, *(const v8u*)(p + k0 + half * 16));
}
__device__ __forceinline__ v8f wmma_bf16(v16bf a, v16bf b, v8f c) {
    return __builtin_amdgcn_wmma_f32_16x16x32_bf16(
        false, a, false, b, (short)0, c, false, false);
}

// ---------------- elementwise helpers ----------------
__global__ void k_f32_to_bf16(const float* __restrict__ in,
                              unsigned short* __restrict__ out, int n) {
    int i = blockIdx.x * blockDim.x + threadIdx.x;
    if (i < n) out[i] = f2bf(in[i]);
}

__global__ void k_deg_init(float* __restrict__ deg, int n) {
    int i = blockIdx.x * blockDim.x + threadIdx.x;
    if (i < n) deg[i] = 1.0f;                 // self-loop contribution
}

__global__ void k_deg_count(const long long* __restrict__ dst,
                            float* __restrict__ deg, int e) {
    int i = blockIdx.x * blockDim.x + threadIdx.x;
    if (i < e) atomicAdd(&deg[dst[i]], 1.0f);
}

__global__ void k_rsqrt_inplace(float* __restrict__ d, int n) {
    int i = blockIdx.x * blockDim.x + threadIdx.x;
    if (i < n) d[i] = rsqrtf(d[i]);           // deg >= 1 always (self-loops)
}

// ---------------- GEMM1: h = X @ W_gcn.T, 2x4 register blocked ----------------
// grid(ceil(N/32), 1) x block(256): wave w covers n-cols [64w, 64w+64),
// rows [32b, 32b+32). 8 WMMA per k-step from 2 A-frags + 4 B-frags.
__global__ void k_gemm1(const unsigned short* __restrict__ Xb,
                        const unsigned short* __restrict__ Wb,
                        float* __restrict__ h) {
    const int wave = threadIdx.x >> 5;        // 0..7 -> n group of 64
    const int lane = threadIdx.x & 31;
    const int half = lane >> 4, l16 = lane & 15;
    const int m0 = blockIdx.x * 32;
    const int m1 = m0 + 16;                   // may be OOB in last block
    const int n0 = wave * 64;

    const unsigned short* __restrict__ a0p = Xb + (size_t)(m0 + l16) * IN_DIM;
    const unsigned short* __restrict__ a1p =
        Xb + (size_t)min(m1 + l16, N_NODES - 1) * IN_DIM;   // clamp, EXEC stays full
    const unsigned short* __restrict__ bbase = Wb + (size_t)(n0 + l16) * IN_DIM;

    v8f acc[2][4] = {};
#pragma unroll 2
    for (int k0 = 0; k0 < IN_DIM; k0 += 32) {
        v16bf a0 = load_afrag(a0p, k0, half);
        v16bf a1 = load_afrag(a1p, k0, half);
#pragma unroll
        for (int j = 0; j < 4; ++j) {
            v16bf bv = load_bfrag(bbase + (size_t)j * 16 * IN_DIM, k0, half);
            acc[0][j] = wmma_bf16(a0, bv, acc[0][j]);
            acc[1][j] = wmma_bf16(a1, bv, acc[1][j]);
        }
    }
    // C/D layout: lane half h, VGPR r -> row (8h + r), col l16
#pragma unroll
    for (int j = 0; j < 4; ++j) {
        float* __restrict__ op =
            h + (size_t)(m0 + half * 8) * HID_DIM + n0 + 16 * j + l16;
#pragma unroll
        for (int r = 0; r < 8; ++r) op[(size_t)r * HID_DIM] = acc[0][j][r];
    }
    if (m1 < N_NODES) {                       // wave-uniform tail skip
#pragma unroll
        for (int j = 0; j < 4; ++j) {
            float* __restrict__ op =
                h + (size_t)(m1 + half * 8) * HID_DIM + n0 + 16 * j + l16;
#pragma unroll
            for (int r = 0; r < 8; ++r) op[(size_t)r * HID_DIM] = acc[1][j][r];
        }
    }
}

// Scatter: one wave per edge; agg[dst] += h[src] * dinv[src]*dinv[dst]
__global__ void k_scatter(const long long* __restrict__ src,
                          const long long* __restrict__ dst,
                          const float* __restrict__ dinv,
                          const float* __restrict__ h,
                          float* __restrict__ agg, int e) {
    const int wid  = (blockIdx.x * blockDim.x + threadIdx.x) >> 5;
    const int lane = threadIdx.x & 31;
    if (wid >= e) return;
    const long long s = src[wid];
    const long long d = dst[wid];
    const float norm  = dinv[s] * dinv[d];
    const float* __restrict__ hs = h + (size_t)s * HID_DIM;
    float* __restrict__ ad       = agg + (size_t)d * HID_DIM;
#pragma unroll
    for (int i = 0; i < HID_DIM / 32; ++i) {
        const int c = lane + 32 * i;
        atomicAdd(ad + c, hs[c] * norm);      // f32 add atomics, L2-resident agg
    }
}

// hidden = relu(agg + h*dinv^2 + b_gcn) -> bf16 (self-loop term fused)
__global__ void k_hidden(const float* __restrict__ agg,
                         const float* __restrict__ h,
                         const float* __restrict__ dinv,
                         const float* __restrict__ bg,
                         unsigned short* __restrict__ hb) {
    const int idx  = blockIdx.x * blockDim.x + threadIdx.x;  // N_NODES*HID_DIM
    const int node = idx >> 9;
    const int kk   = idx & (HID_DIM - 1);
    const float di = dinv[node];
    float v = agg[idx] + h[idx] * di * di + bg[kk];
    hb[idx] = f2bf(fmaxf(v, 0.0f));
}

// ---------------- GEMM2: out = hidden @ W_cls.T + b_cls, 2x4 blocked ----------
// Wave-linear: global wave gw covers rows [32gw, 32gw+32), all 4 n-tiles (64 cols).
__global__ void k_gemm2(const unsigned short* __restrict__ Hb,
                        const unsigned short* __restrict__ Wcb,
                        const float* __restrict__ bcls,
                        float* __restrict__ out) {
    const int gw   = (blockIdx.x * blockDim.x + threadIdx.x) >> 5;
    const int lane = threadIdx.x & 31;
    const int half = lane >> 4, l16 = lane & 15;
    const int m0 = gw * 32;
    if (m0 >= N_NODES) return;                // uniform: wave runs no WMMA
    const int m1 = m0 + 16;

    const unsigned short* __restrict__ a0p = Hb + (size_t)(m0 + l16) * HID_DIM;
    const unsigned short* __restrict__ a1p =
        Hb + (size_t)min(m1 + l16, N_NODES - 1) * HID_DIM;
    const unsigned short* __restrict__ bbase = Wcb + (size_t)l16 * HID_DIM;

    v8f acc[2][4] = {};
#pragma unroll 2
    for (int k0 = 0; k0 < HID_DIM; k0 += 32) {
        v16bf a0 = load_afrag(a0p, k0, half);
        v16bf a1 = load_afrag(a1p, k0, half);
#pragma unroll
        for (int j = 0; j < 4; ++j) {
            v16bf bv = load_bfrag(bbase + (size_t)j * 16 * HID_DIM, k0, half);
            acc[0][j] = wmma_bf16(a0, bv, acc[0][j]);
            acc[1][j] = wmma_bf16(a1, bv, acc[1][j]);
        }
    }
#pragma unroll
    for (int j = 0; j < 4; ++j) {
        const float bias = bcls[16 * j + l16];
        float* __restrict__ op =
            out + (size_t)(m0 + half * 8) * N_CLASSES + 16 * j + l16;
#pragma unroll
        for (int r = 0; r < 8; ++r) op[(size_t)r * N_CLASSES] = acc[0][j][r] + bias;
    }
    if (m1 < N_NODES) {
#pragma unroll
        for (int j = 0; j < 4; ++j) {
            const float bias = bcls[16 * j + l16];
            float* __restrict__ op =
                out + (size_t)(m1 + half * 8) * N_CLASSES + 16 * j + l16;
#pragma unroll
            for (int r = 0; r < 8; ++r) op[(size_t)r * N_CLASSES] = acc[1][j][r] + bias;
        }
    }
}

// ---------------------------------------------------------------------------
extern "C" void kernel_launch(void* const* d_in, const int* in_sizes, int n_in,
                              void* d_out, int out_size, void* d_ws, size_t ws_size,
                              hipStream_t stream) {
    const float*     X    = (const float*)d_in[0];
    const long long* ei   = (const long long*)d_in[1];   // [2, E] int64
    const float*     Wg   = (const float*)d_in[2];       // [HID, IN]
    const float*     bg   = (const float*)d_in[3];       // [HID]
    const float*     Wc   = (const float*)d_in[4];       // [CLS, HID]
    const float*     bc   = (const float*)d_in[5];       // [CLS]
    float*           out  = (float*)d_out;

    const long long* srcIdx = ei;
    const long long* dstIdx = ei + N_EDGES;

    // ---- workspace layout (bytes) ----
    char* ws = (char*)d_ws;
    size_t off = 0;
    float* dinv = (float*)(ws + off);  off += ((size_t)N_NODES * 4 + 255) & ~(size_t)255;
    unsigned short* Wb  = (unsigned short*)(ws + off); off += (size_t)HID_DIM * IN_DIM * 2;
    unsigned short* Wcb = (unsigned short*)(ws + off); off += (size_t)N_CLASSES * HID_DIM * 2;
    unsigned short* Xb  = (unsigned short*)(ws + off); off += (size_t)N_NODES * IN_DIM * 2;
    float* h   = (float*)(ws + off);   off += (size_t)N_NODES * HID_DIM * 4;
    float* agg = (float*)(ws + off);   off += (size_t)N_NODES * HID_DIM * 4;
    unsigned short* Hb = Xb;           // reuse: Xb dead after GEMM1

    // ---- degrees / normalization ----
    k_deg_init<<<(N_NODES + 255) / 256, 256, 0, stream>>>(dinv, N_NODES);
    k_deg_count<<<(N_EDGES + 255) / 256, 256, 0, stream>>>(dstIdx, dinv, N_EDGES);
    k_rsqrt_inplace<<<(N_NODES + 255) / 256, 256, 0, stream>>>(dinv, N_NODES);

    // ---- bf16 conversions ----
    k_f32_to_bf16<<<(N_NODES * IN_DIM) / 256, 256, 0, stream>>>(X, Xb, N_NODES * IN_DIM);
    k_f32_to_bf16<<<(HID_DIM * IN_DIM) / 256, 256, 0, stream>>>(Wg, Wb, HID_DIM * IN_DIM);
    k_f32_to_bf16<<<(N_CLASSES * HID_DIM) / 256, 256, 0, stream>>>(Wc, Wcb, N_CLASSES * HID_DIM);

    // ---- h = X @ W_gcn.T (WMMA, 2x4 blocked) ----
    k_gemm1<<<(N_NODES + 31) / 32, 256, 0, stream>>>(Xb, Wb, h);

    // ---- agg = 0; scatter edges ----
    hipMemsetAsync(agg, 0, (size_t)N_NODES * HID_DIM * 4, stream);
    k_scatter<<<(N_EDGES + 7) / 8, 256, 0, stream>>>(srcIdx, dstIdx, dinv, h, agg, N_EDGES);

    // ---- hidden = relu(agg + selfloop + b) -> bf16 ----
    k_hidden<<<(N_NODES * HID_DIM) / 256, 256, 0, stream>>>(agg, h, dinv, bg, Hb);

    // ---- logits = hidden @ W_cls.T + b_cls (WMMA, 2x4 blocked) ----
    {
        int waves  = (N_NODES + 31) / 32;           // 1563
        int blocks = (waves * 32 + 255) / 256;      // 196 blocks of 8 waves
        k_gemm2<<<blocks, 256, 0, stream>>>(Hb, Wcb, bc, out);
    }

    (void)in_sizes; (void)n_in; (void)out_size; (void)ws_size;
}
